// PermutohedralX_65120294142165
// MI455X (gfx1250) — matrix operand order repaired and checked by reading, not compiled, not run on hardware
//
#include <hip/hip_runtime.h>
#include <hip/hip_bf16.h>

typedef __attribute__((ext_vector_type(16))) _Float16 v16h;
typedef __attribute__((ext_vector_type(8)))  float    v8f;

constexpr int D1 = 6;   // d+1 lattice vertices per point
constexpr int VS = 8;   // value vector size

// ---------------- zero scratch (both ping-pong value tables) ----------------
__global__ void pl_zero(float4* __restrict__ p, long n4) {
  long i = (long)blockIdx.x * blockDim.x + threadIdx.x;
  if (i < n4) p[i] = make_float4(0.f, 0.f, 0.f, 0.f);
}

// ---------------- splat: scatter-add into L2-resident value table ----------------
__global__ void pl_splat(const float* __restrict__ inp, const float* __restrict__ ws,
                         const int* __restrict__ os, float* __restrict__ vals, int N) {
  int n = blockIdx.x * blockDim.x + threadIdx.x;
  if (n >= N) return;
  const float4* ip = (const float4*)(inp + (size_t)n * VS);
  float4 v0 = ip[0], v1 = ip[1];
#pragma unroll
  for (int k = 0; k < D1; ++k) {
    float w = ws[(size_t)n * D1 + k];
    float* b = vals + (size_t)(os[(size_t)n * D1 + k] + 1) * VS;
    atomicAdd(b + 0, v0.x * w);
    atomicAdd(b + 1, v0.y * w);
    atomicAdd(b + 2, v0.z * w);
    atomicAdd(b + 3, v0.w * w);
    atomicAdd(b + 4, v1.x * w);
    atomicAdd(b + 5, v1.y * w);
    atomicAdd(b + 6, v1.z * w);
    atomicAdd(b + 7, v1.w * w);
  }
}

// ---------------- blur: vout[i+1] = vin[i+1] + 0.5*(vin[a] + vin[b]) ----------------
// Contiguous center-row stream staged through LDS with the CDNA5 async-copy path
// (GLOBAL_LOAD_ASYNC_TO_LDS_B128 / ASYNCcnt); random neighbor gathers go straight to
// VGPRs (they hit L2: the 32 MB table << 192 MB L2).
__global__ void pl_blur(const float* __restrict__ vin, float* __restrict__ vout,
                        const int* __restrict__ bnj, int M) {
  __shared__ __align__(16) float rowbuf[256 * VS];
  int t = threadIdx.x;
  long i = (long)blockIdx.x * 256 + t;
  if (i >= M) return;

  const float* src = vin + (size_t)(i + 1) * VS;
  unsigned lds0 = (unsigned)(size_t)&rowbuf[t * VS];
  unsigned long long ga = (unsigned long long)(size_t)src;
  // INST_OFFSET is added on both LDS and memory side, so one base covers both halves.
  asm volatile("global_load_async_to_lds_b128 %0, %1, off"
               :: "v"(lds0), "v"(ga) : "memory");
  asm volatile("global_load_async_to_lds_b128 %0, %1, off offset:16"
               :: "v"(lds0), "v"(ga) : "memory");

  int2 nb = ((const int2*)bnj)[i];              // neighbor rows in [0, M]; 0 = zero pad row
  const float4* pa = (const float4*)(vin + (size_t)nb.x * VS);
  const float4* pb = (const float4*)(vin + (size_t)nb.y * VS);
  float4 a0 = pa[0], a1 = pa[1];
  float4 b0 = pb[0], b1 = pb[1];

  asm volatile("s_wait_asynccnt 0" ::: "memory");
  const float4* cp = (const float4*)&rowbuf[t * VS];
  float4 c0 = cp[0], c1 = cp[1];

  float4 o0, o1;
  o0.x = fmaf(0.5f, a0.x + b0.x, c0.x);
  o0.y = fmaf(0.5f, a0.y + b0.y, c0.y);
  o0.z = fmaf(0.5f, a0.z + b0.z, c0.z);
  o0.w = fmaf(0.5f, a0.w + b0.w, c0.w);
  o1.x = fmaf(0.5f, a1.x + b1.x, c1.x);
  o1.y = fmaf(0.5f, a1.y + b1.y, c1.y);
  o1.z = fmaf(0.5f, a1.z + b1.z, c1.z);
  o1.w = fmaf(0.5f, a1.w + b1.w, c1.w);
  float4* op = (float4*)(vout + (size_t)(i + 1) * VS);
  op[0] = o0; op[1] = o1;
}

// ---------------- slice: einsum('nk,nkv->nv') as block-diagonal WMMA GEMM ----------------
// One wave per 16-point tile. Pair index p = n*6+k (96 pairs/tile) is the K dimension:
//   A[16x96] block-diag:  A[n][p] = ws[n][k] iff p belongs to point n
//   B[96x16]:             B[p][v] = values[os[n][k]+1][v]  (cols 8..15 structurally zero)
// 3 chunks of K=32 via v_wmma_f32_16x16x32_f16, with f32->f16 hi/lo splitting of BOTH
// operands (AhBh + AhBl + AlBh) for near-f32 accuracy.
//
// B is staged in LDS in *fragment order*: g[c][lane][e] = B[32c + (lane&16) + e][lane&15],
// so each reader lane fills its whole v16h with two ds_load_b128 (32 contiguous bytes).
// Lanes owning B columns 8..15 (all-zero) skip LDS entirely and feed a zero vector.
__global__ void pl_slice_wmma(const float* __restrict__ ws, const int* __restrict__ os,
                              const float* __restrict__ vals, float* __restrict__ out,
                              float alpha) {
  __shared__ __align__(16) _Float16 gh[4][3][32][16];   // hi halves, fragment-ordered
  __shared__ __align__(16) _Float16 gl[4][3][32][16];   // lo halves, fragment-ordered
  const int lane = threadIdx.x & 31;
  const int wv   = threadIdx.x >> 5;
  const long base = ((long)blockIdx.x * 4 + wv) * 16;   // first point of this tile
  const int m   = lane & 15;                            // A row and B/D column owned by lane
  const int grp = lane & 16;                            // K-half group (0 or 16)

  // Per-lane weight row (A operand source), split hi/lo.
  _Float16 whi[D1], wlo[D1];
#pragma unroll
  for (int k = 0; k < D1; ++k) {
    float w = ws[(base + m) * D1 + k];
    _Float16 h = (_Float16)w;
    whi[k] = h;
    wlo[k] = (_Float16)(w - (float)h);
  }

  // Gather the tile's 96 lattice rows; scatter into fragment-ordered LDS (hi/lo).
  // Lane t handles K-row p = t + 32c; its value for column v lands in slot
  // g[c][(t&16)+v][t&15].
#pragma unroll
  for (int c = 0; c < 3; ++c) {
    int p = lane + 32 * c;
    int n = p / D1, k = p % D1;
    long row = (long)os[(base + n) * D1 + k] + 1;
    const float4* g = (const float4*)(vals + row * VS);
    float4 g0 = g[0], g1 = g[1];
    float gv[VS] = {g0.x, g0.y, g0.z, g0.w, g1.x, g1.y, g1.z, g1.w};
#pragma unroll
    for (int v = 0; v < VS; ++v) {
      _Float16 h = (_Float16)gv[v];
      gh[wv][c][grp + v][m] = h;
      gl[wv][c][grp + v][m] = (_Float16)(gv[v] - (float)h);
    }
  }
  __syncthreads();

  v8f acc = {};
  const v16h zf = {};
#pragma unroll
  for (int c = 0; c < 3; ++c) {
    // A fragment, contiguous K order (same as B in LDS): lanes 0-15 hold K 0-15, lanes 16-31 hold K 16-31.
    v16h ah, al;
    if (lane < 16) {
#pragma unroll
      for (int e = 0; e < 16; ++e) {
        const int kk = e;
        const int p = 32 * c + kk;
        const int n = p / D1, k = p % D1;
        bool mine = (n == m);
        ah[e] = mine ? whi[k] : (_Float16)0.f;
        al[e] = mine ? wlo[k] : (_Float16)0.f;
      }
    } else {
#pragma unroll
      for (int e = 0; e < 16; ++e) {
        const int kk = 16 + e;
        const int p = 32 * c + kk;
        const int n = p / D1, k = p % D1;
        bool mine = (n == m);
        ah[e] = mine ? whi[k] : (_Float16)0.f;
        al[e] = mine ? wlo[k] : (_Float16)0.f;
      }
    }
    // B fragment: whole v16h in two ds_load_b128; zero vector for the all-zero columns.
    v16h bhf, blf;
    if (m < VS) {
      bhf = *(const v16h*)&gh[wv][c][lane][0];
      blf = *(const v16h*)&gl[wv][c][lane][0];
    } else {
      bhf = zf;
      blf = zf;
    }
    acc = __builtin_amdgcn_wmma_f32_16x16x32_f16(false, ah, false, bhf, (short)0, acc, false, false);
    acc = __builtin_amdgcn_wmma_f32_16x16x32_f16(false, ah, false, blf, (short)0, acc, false, false);
    acc = __builtin_amdgcn_wmma_f32_16x16x32_f16(false, al, false, bhf, (short)0, acc, false, false);
  }

  // D layout (32-bit 16x16): lane holds rows rb..rb+7 of column m.
  const int rb = (lane < 16) ? 0 : 8;
  if (m < VS) {
#pragma unroll
    for (int r = 0; r < 8; ++r)
      out[(base + rb + r) * VS + m] = alpha * acc[r];
  }
}

// ---------------- scalar slice for any tail points (N % 64 != 0) ----------------
__global__ void pl_slice_tail(const float* __restrict__ ws, const int* __restrict__ os,
                              const float* __restrict__ vals, float* __restrict__ out,
                              int start, int N, float alpha) {
  int n = start + blockIdx.x * blockDim.x + threadIdx.x;
  if (n >= N) return;
  float acc[VS] = {};
#pragma unroll
  for (int k = 0; k < D1; ++k) {
    float w = ws[(size_t)n * D1 + k];
    long row = (long)os[(size_t)n * D1 + k] + 1;
    const float4* g = (const float4*)(vals + row * VS);
    float4 g0 = g[0], g1 = g[1];
    acc[0] += w * g0.x; acc[1] += w * g0.y; acc[2] += w * g0.z; acc[3] += w * g0.w;
    acc[4] += w * g1.x; acc[5] += w * g1.y; acc[6] += w * g1.z; acc[7] += w * g1.w;
  }
  float4* o = (float4*)(out + (size_t)n * VS);
  o[0] = make_float4(alpha * acc[0], alpha * acc[1], alpha * acc[2], alpha * acc[3]);
  o[1] = make_float4(alpha * acc[4], alpha * acc[5], alpha * acc[6], alpha * acc[7]);
}

extern "C" void kernel_launch(void* const* d_in, const int* in_sizes, int n_in,
                              void* d_out, int out_size, void* d_ws, size_t ws_size,
                              hipStream_t stream) {
  (void)n_in; (void)out_size; (void)ws_size;
  const float* inp = (const float*)d_in[0];
  const float* ws  = (const float*)d_in[1];
  const int*   os  = (const int*)d_in[2];
  const int*   bn  = (const int*)d_in[3];
  float* out = (float*)d_out;

  const int N = in_sizes[0] / VS;           // 1048576
  const int M = in_sizes[3] / (D1 * 2);     // 1000000
  const float alpha = 1.0f / (1.0f + 1.0f / (float)(1 << (D1 - 1)));

  float* valsA = (float*)d_ws;                       // (M+1) x VS
  float* valsB = valsA + (size_t)(M + 1) * VS;       // (M+1) x VS  (64 MB total, L2-resident)

  // Zero both value tables (row 0 = padding row must stay zero; replays re-use scratch).
  long n4 = (2L * (M + 1) * VS) / 4;                 // 2*(M+1)*VS is a multiple of 16
  pl_zero<<<(int)((n4 + 255) / 256), 256, 0, stream>>>((float4*)d_ws, n4);

  pl_splat<<<(N + 255) / 256, 256, 0, stream>>>(inp, ws, os, valsA, N);

  for (int j = 0; j < D1; ++j) {                     // ping-pong: gathers must see pre-update values
    const float* vi = (j & 1) ? valsB : valsA;
    float*       vo = (j & 1) ? valsA : valsB;
    pl_blur<<<(int)(((long)M + 255) / 256), 256, 0, stream>>>(vi, vo, bn + (size_t)j * M * 2, M);
  }
  const float* vfin = (D1 & 1) ? valsB : valsA;      // D1 even -> valsA

  int blocks = N / 64;                               // 4 tiles (waves) of 16 points per block
  if (blocks > 0)
    pl_slice_wmma<<<blocks, 128, 0, stream>>>(ws, os, vfin, out, alpha);
  int done = blocks * 64;
  if (N - done > 0)
    pl_slice_tail<<<(N - done + 255) / 256, 256, 0, stream>>>(ws, os, vfin, out, done, N, alpha);
}
